// GSVectorQuantizer_87041807220990
// MI455X (gfx1250) — matrix operand order repaired and checked by reading, not compiled, not run on hardware
//
#include <hip/hip_runtime.h>
#include <hip/hip_bf16.h>
#include <math.h>

// GS vector quantizer, fused, bf16 WMMA path for gfx1250 (MI455X).
// TILE_M=32 rows per workgroup (16 waves) halves L2 traffic on the bf16
// weight operands; gumbel logits are spilled to a lane-major LDS region
// instead of registers to keep the GEMM loops spill-free.
// Outputs (concatenated): quantized [B*D] f32, loss [1] f32, indices [B] (as f32).

typedef __attribute__((ext_vector_type(16))) __bf16 v16bf;
typedef __attribute__((ext_vector_type(8)))  __bf16 v8bf;
typedef __attribute__((ext_vector_type(8)))  float  v8f;

#define B_ROWS 16384
#define D_DIM  512
#define K_DIM  1024
#define TILE_M 32
#define WAVES  16
#define TPB    (WAVES * 32)
#define TEMP_INV 2.0f              // 1 / 0.5
#define LOG_K  6.9314718055994531f // log(1024)

__device__ __forceinline__ __bf16 f2bf(float f) { return (__bf16)f; }

// ---------------- prep: W -> bf16 (row-major + transposed), wsq, zero loss ----
__global__ void gsvq_prep(const float* __restrict__ weight,
                          __bf16* __restrict__ wbf, __bf16* __restrict__ wtbf,
                          float* __restrict__ wsq, float* __restrict__ lossAcc) {
    const int k = blockIdx.x;
    const int tid = threadIdx.x;
    if (k == 0 && tid == 0) *lossAcc = 0.0f;
    __shared__ float red[256];
    float ss = 0.0f;
    for (int d = tid; d < D_DIM; d += 256) {
        float w = weight[(size_t)k * D_DIM + d];
        __bf16 wb = f2bf(w);
        wbf [(size_t)k * D_DIM + d] = wb;  // [K][D]
        wtbf[(size_t)d * K_DIM + k] = wb;  // [D][K]
        ss += w * w;
    }
    red[tid] = ss;
    __syncthreads();
    for (int s = 128; s > 0; s >>= 1) {
        if (tid < s) red[tid] += red[tid + s];
        __syncthreads();
    }
    if (tid == 0) wsq[k] = red[0];
}

// ---------------- main fused kernel -----------------------------------------
__global__ __launch_bounds__(TPB)
void gsvq_main(const float* __restrict__ x, const float* __restrict__ u,
               const __bf16* __restrict__ wbf, const __bf16* __restrict__ wtbf,
               const float* __restrict__ wsq, float* __restrict__ out,
               float* __restrict__ lossAcc) {
    __shared__ __bf16 xs[TILE_M * D_DIM];   // 32 KB: x tile (A of GEMM1)
    __shared__ __bf16 Pb[TILE_M * K_DIM];   // 64 KB: sample probs (A of GEMM2)
    __shared__ float  TLs[64 * TPB];        // 128 KB: per-lane gumbel-logit spill
    __shared__ float rMaxV[TILE_M][WAVES];  // 12 KB reductions total
    __shared__ int   rMaxI[TILE_M][WAVES];
    __shared__ float rMaxT[TILE_M][WAVES];
    __shared__ float rS [TILE_M][WAVES];
    __shared__ float rWS[TILE_M][WAVES];
    __shared__ float rS2[TILE_M][WAVES];

    const int tid  = threadIdx.x;
    const int w    = tid >> 5;      // wave 0..15
    const int lane = tid & 31;
    const int l16  = lane & 15;
    const int half = lane >> 4;
    const int rowBase = blockIdx.x * TILE_M;

    // lane-private TL slot accessor: lane-major layout -> bank-conflict free
#define TL_AT(t, mt, v) TLs[((((t) << 1) | (mt)) * 8 + (v)) * TPB + tid]

    // stage x tile -> bf16 LDS
    for (int i = tid; i < TILE_M * D_DIM; i += TPB)
        xs[i] = f2bf(x[(size_t)rowBase * D_DIM + i]);
    __syncthreads();

    // ---- GEMM1: logits = 2*x.Wt - |w|^2 ----
    // wave owns codes [64w, 64w+64) (4 N-tiles), rows: 2 M-tiles of 16.
    v8f zero = {};
    v8f acc[4][2];
#pragma unroll
    for (int t = 0; t < 4; ++t)
#pragma unroll
        for (int mt = 0; mt < 2; ++mt) acc[t][mt] = zero;

#pragma unroll 1
    for (int d0 = 0; d0 < D_DIM; d0 += 32) {
        v16bf a[2];
#pragma unroll
        for (int mt = 0; mt < 2; ++mt) {
            const int mrow = 16 * mt + l16;
            v8bf alo = *(const v8bf*)&xs[mrow * D_DIM + d0 + 8 * half];
            v8bf ahi = *(const v8bf*)&xs[mrow * D_DIM + d0 + 16 + 8 * half];
            a[mt] = __builtin_shufflevector(alo, ahi, 0,1,2,3,4,5,6,7,8,9,10,11,12,13,14,15);
        }
#pragma unroll
        for (int t = 0; t < 4; ++t) {
            const int n = (w << 6) + (t << 4) + l16;
            v8bf blo = *(const v8bf*)&wbf[(size_t)n * D_DIM + d0 + 8 * half];
            v8bf bhi = *(const v8bf*)&wbf[(size_t)n * D_DIM + d0 + 16 + 8 * half];
            v16bf b = __builtin_shufflevector(blo, bhi, 0,1,2,3,4,5,6,7,8,9,10,11,12,13,14,15);
#pragma unroll
            for (int mt = 0; mt < 2; ++mt)
                acc[t][mt] = __builtin_amdgcn_wmma_f32_16x16x32_bf16(
                    false, a[mt], false, b, (short)0, acc[t][mt], false, false);
        }
    }

    // effective logits (row-constant |x|^2 cancels) + gumbel logits -> LDS
#pragma unroll
    for (int t = 0; t < 4; ++t) {
        const int n = (w << 6) + (t << 4) + l16;
        const float wq = wsq[n];
#pragma unroll
        for (int mt = 0; mt < 2; ++mt) {
#pragma unroll
            for (int v = 0; v < 8; ++v) {
                const float lg = 2.0f * acc[t][mt][v] - wq;   // effective logit
                acc[t][mt][v] = lg;                           // keep L in acc
                const int r = 16 * mt + v + 8 * half;
                const float uu = u[(size_t)(rowBase + r) * K_DIM + n];
                const float g = -__logf(-__logf(uu));
                TL_AT(t, mt, v) = (lg + g) * TEMP_INV;
            }
        }
    }

    // ---- round 1: row max / argmax (L), row max (gumbel) ----
    float fm[2][8], ft[2][8]; int fi[2][8];
#pragma unroll
    for (int mt = 0; mt < 2; ++mt) {
#pragma unroll
        for (int v = 0; v < 8; ++v) {
            float bv = -INFINITY, bt = -INFINITY; int bi = 0;
#pragma unroll
            for (int t = 0; t < 4; ++t) {
                const int n = (w << 6) + (t << 4) + l16;
                if (acc[t][mt][v] > bv) { bv = acc[t][mt][v]; bi = n; }
                bt = fmaxf(bt, TL_AT(t, mt, v));
            }
            for (int o = 1; o < 16; o <<= 1) {
                float ov = __shfl_xor(bv, o, 16);
                int   oi = __shfl_xor(bi, o, 16);
                float ot = __shfl_xor(bt, o, 16);
                if (ov > bv || (ov == bv && oi < bi)) { bv = ov; bi = oi; }
                bt = fmaxf(bt, ot);
            }
            if (l16 == 0) {
                const int r = 16 * mt + v + 8 * half;
                rMaxV[r][w] = bv; rMaxI[r][w] = bi; rMaxT[r][w] = bt;
            }
        }
    }
    __syncthreads();
#pragma unroll
    for (int mt = 0; mt < 2; ++mt) {
#pragma unroll
        for (int v = 0; v < 8; ++v) {
            const int r = 16 * mt + v + 8 * half;
            float bv = -INFINITY, bt = -INFINITY; int bi = 0;
#pragma unroll
            for (int q = 0; q < WAVES; ++q) {
                float ov = rMaxV[r][q]; int oi = rMaxI[r][q]; float ot = rMaxT[r][q];
                if (ov > bv || (ov == bv && oi < bi)) { bv = ov; bi = oi; }
                bt = fmaxf(bt, ot);
            }
            fm[mt][v] = bv; fi[mt][v] = bi; ft[mt][v] = bt;
        }
    }

    // ---- round 2: sumexp, weighted sum (KL), gumbel sumexp ----
    float sE[2][8], sW[2][8], sT[2][8];
#pragma unroll
    for (int mt = 0; mt < 2; ++mt) {
#pragma unroll
        for (int v = 0; v < 8; ++v) {
            float se = 0.0f, swm = 0.0f, st = 0.0f;
#pragma unroll
            for (int t = 0; t < 4; ++t) {
                const float d = acc[t][mt][v] - fm[mt][v];
                const float e = __expf(d);
                se += e; swm += d * e;
                st += __expf(TL_AT(t, mt, v) - ft[mt][v]);
            }
            for (int o = 1; o < 16; o <<= 1) {
                se  += __shfl_xor(se,  o, 16);
                swm += __shfl_xor(swm, o, 16);
                st  += __shfl_xor(st,  o, 16);
            }
            if (l16 == 0) {
                const int r = 16 * mt + v + 8 * half;
                rS[r][w] = se; rWS[r][w] = swm; rS2[r][w] = st;
            }
        }
    }
    __syncthreads();
#pragma unroll
    for (int mt = 0; mt < 2; ++mt) {
#pragma unroll
        for (int v = 0; v < 8; ++v) {
            const int r = 16 * mt + v + 8 * half;
            float se = 0.0f, swm = 0.0f, st = 0.0f;
#pragma unroll
            for (int q = 0; q < WAVES; ++q) { se += rS[r][q]; swm += rWS[r][q]; st += rS2[r][q]; }
            sE[mt][v] = se; sW[mt][v] = swm; sT[mt][v] = st;
        }
    }

    // loss (mean KL) + argmin indices, once per row (wave 0, one lane per half)
    if (w == 0 && l16 == 0) {
        float klsum = 0.0f;
#pragma unroll
        for (int mt = 0; mt < 2; ++mt) {
#pragma unroll
            for (int v = 0; v < 8; ++v) {
                const int r = 16 * mt + v + 8 * half;
                klsum += sW[mt][v] / sE[mt][v] - __logf(sE[mt][v]) + LOG_K;
                out[(size_t)B_ROWS * D_DIM + 1 + rowBase + r] = (float)fi[mt][v];
            }
        }
        atomicAdd(lossAcc, klsum * (1.0f / (float)B_ROWS));
    }

    // sample probs -> LDS bf16 (A-matrix for GEMM2)
#pragma unroll
    for (int t = 0; t < 4; ++t) {
        const int n = (w << 6) + (t << 4) + l16;
#pragma unroll
        for (int mt = 0; mt < 2; ++mt) {
#pragma unroll
            for (int v = 0; v < 8; ++v) {
                const int r = 16 * mt + v + 8 * half;
                const float p = __expf(TL_AT(t, mt, v) - ft[mt][v]) / sT[mt][v];
                Pb[r * K_DIM + n] = f2bf(p);
            }
        }
    }
    __syncthreads();

    // ---- GEMM2: quantized = P @ W ; wave owns cols [32w, 32w+32) ----
    v8f acc2[2][2];
#pragma unroll
    for (int td = 0; td < 2; ++td)
#pragma unroll
        for (int mt = 0; mt < 2; ++mt) acc2[td][mt] = zero;

#pragma unroll 1
    for (int k0 = 0; k0 < K_DIM; k0 += 32) {
        v16bf a[2];
#pragma unroll
        for (int mt = 0; mt < 2; ++mt) {
            const int mrow = 16 * mt + l16;
            v8bf alo = *(const v8bf*)&Pb[mrow * K_DIM + k0 + 8 * half];
            v8bf ahi = *(const v8bf*)&Pb[mrow * K_DIM + k0 + 16 + 8 * half];
            a[mt] = __builtin_shufflevector(alo, ahi, 0,1,2,3,4,5,6,7,8,9,10,11,12,13,14,15);
        }
#pragma unroll
        for (int td = 0; td < 2; ++td) {
            const int d0 = (w << 5) + (td << 4);
            v8bf blo = *(const v8bf*)&wtbf[(size_t)(d0 + l16) * K_DIM + k0 + 8 * half];
            v8bf bhi = *(const v8bf*)&wtbf[(size_t)(d0 + l16) * K_DIM + k0 + 16 + 8 * half];
            v16bf b = __builtin_shufflevector(blo, bhi, 0,1,2,3,4,5,6,7,8,9,10,11,12,13,14,15);
#pragma unroll
            for (int mt = 0; mt < 2; ++mt)
                acc2[td][mt] = __builtin_amdgcn_wmma_f32_16x16x32_bf16(
                    false, a[mt], false, b, (short)0, acc2[td][mt], false, false);
        }
    }

    // store quantized (C layout: N across lanes, M across VGPRs)
#pragma unroll
    for (int td = 0; td < 2; ++td) {
        const int d0 = (w << 5) + (td << 4);
#pragma unroll
        for (int mt = 0; mt < 2; ++mt) {
#pragma unroll
            for (int v = 0; v < 8; ++v) {
                const int r = 16 * mt + v + 8 * half;
                out[(size_t)(rowBase + r) * D_DIM + d0 + l16] = acc2[td][mt][v];
            }
        }
    }
#undef TL_AT
}

// ---------------- finalize loss ---------------------------------------------
__global__ void gsvq_finalize(const float* __restrict__ lossAcc, float* __restrict__ out) {
    if (threadIdx.x == 0 && blockIdx.x == 0)
        out[(size_t)B_ROWS * D_DIM] = *lossAcc;
}

extern "C" void kernel_launch(void* const* d_in, const int* in_sizes, int n_in,
                              void* d_out, int out_size, void* d_ws, size_t ws_size,
                              hipStream_t stream) {
    (void)in_sizes; (void)n_in; (void)out_size; (void)ws_size;
    const float* x      = (const float*)d_in[0];
    // d_in[1] = var (unused by reference quantize path)
    const float* weight = (const float*)d_in[2];
    const float* u      = (const float*)d_in[3];
    float* out = (float*)d_out;

    char* ws = (char*)d_ws;
    __bf16* wbf  = (__bf16*)ws;                                    // 1 MB  [K][D] bf16
    __bf16* wtbf = (__bf16*)(ws + (size_t)K_DIM * D_DIM * 2);      // 1 MB  [D][K] bf16
    float*  wsq  = (float*) (ws + (size_t)K_DIM * D_DIM * 4);      // 4 KB
    float*  lossAcc = (float*)(ws + (size_t)K_DIM * D_DIM * 4 + K_DIM * 4);

    gsvq_prep<<<K_DIM, 256, 0, stream>>>(weight, wbf, wtbf, wsq, lossAcc);
    gsvq_main<<<B_ROWS / TILE_M, TPB, 0, stream>>>(x, u, wbf, wtbf, wsq, out, lossAcc);
    gsvq_finalize<<<1, 1, 0, stream>>>(lossAcc, out);
}
